// GAT_GraphConv_24592982737488
// MI455X (gfx1250) — matrix operand order repaired and checked by reading, not compiled, not run on hardware
//
#include <hip/hip_runtime.h>
#include <hip/hip_bf16.h>
#include <math.h>

// ---------------- problem constants ----------------
#define NN   50000
#define EE   400000
#define ETOT (EE + NN)      // 450000 edges incl. self loops
#define INF_ 128
#define HH   8
#define CC   32
#define HC   256            // H*C
#define ED_  16
#define NEG_SLOPE 0.2f
#define LN_EPS 1e-5f
#define EDGE_BLOCKS 1024    // grid-stride blocks for edge_logits

typedef __attribute__((ext_vector_type(16))) __bf16 v16bf;
typedef __attribute__((ext_vector_type(8)))  float  v8f;

// ---------------- small helpers ----------------
__device__ __forceinline__ float atomicMaxFloat(float* addr, float value) {
  int* ia = (int*)addr;
  int old = __float_as_int(*addr);
  while (__int_as_float(old) < value) {
    int assumed = old;
    old = atomicCAS(ia, assumed, __float_as_int(value));
    if (old == assumed) break;
  }
  return __int_as_float(old);
}

__global__ void fillk(float* p, float v, int n) {
  int i = blockIdx.x * blockDim.x + threadIdx.x;
  if (i < n) p[i] = v;
}

// ---------------- self-loop attr = mean of incoming edge attrs ----------------
__global__ void edge_accum(const int* __restrict__ dstA, const float* __restrict__ ea,
                           float* __restrict__ deg, float* __restrict__ loop_attr) {
  int e = blockIdx.x * blockDim.x + threadIdx.x;
  if (e >= EE) return;
  int d = dstA[e];
  atomicAdd(&deg[d], 1.0f);
  const float* a = ea + (size_t)e * ED_;
  float* lp = loop_attr + (size_t)d * ED_;
#pragma unroll
  for (int i = 0; i < ED_; ++i) atomicAdd(&lp[i], a[i]);
}

__global__ void loop_norm(float* __restrict__ loop_attr, const float* __restrict__ deg) {
  int i = blockIdx.x * blockDim.x + threadIdx.x;
  if (i >= NN * ED_) return;
  loop_attr[i] = loop_attr[i] / fmaxf(deg[i >> 4], 1.0f);
}

// ---------------- WMMA dual GEMM: C0 = A@B0+bias0, C1 = A@B1+bias1 ----------------
// A: MxK f32 row-major (M multiple of 16, K multiple of 32), B: KxHC, C: MxHC.
// grid = (HC/128, M/16), block = 256 (8 waves, each owns one 16-col tile).
// A tile staged in LDS once per block (shared by all 8 waves), padded vs bank conflicts.
__global__ void wmma_dual_gemm(const float* __restrict__ A, int K,
                               const float* __restrict__ B0, const float* __restrict__ bias0,
                               float* __restrict__ C0,
                               const float* __restrict__ B1, const float* __restrict__ bias1,
                               float* __restrict__ C1) {
  __shared__ float sA[16 * 33];                    // 16 rows x 32 cols, stride 33 (no conflicts)
  const int tid  = threadIdx.x;
  const int lane = tid & 31;
  const int wv   = tid >> 5;
  const int lo   = lane & 15;
  const int hi   = lane >> 4;
  const int n0   = (blockIdx.x * 8 + wv) * 16;
  const int m0   = blockIdx.y * 16;

  v8f c0 = {}; v8f c1 = {};

  for (int kt = 0; kt < K; kt += 32) {
    __syncthreads();
    {
      // cooperative A-tile stage: 512 floats, 2 per thread, 128B/wave coalesced
      int i0 = tid;                                // row = wv, col = lane
      sA[(i0 >> 5) * 33 + (i0 & 31)] = A[(size_t)(m0 + (i0 >> 5)) * K + kt + (i0 & 31)];
      int i1 = tid + 256;
      sA[(i1 >> 5) * 33 + (i1 & 31)] = A[(size_t)(m0 + (i1 >> 5)) * K + kt + (i1 & 31)];
    }
    __syncthreads();

    v16bf a, b0, b1;
#pragma unroll
    for (int v = 0; v < 8; ++v) {
#pragma unroll
      for (int hf = 0; hf < 2; ++hf) {
        const int e  = 2 * v + hf;
        // A 16x32 bf16 layout: lanes 0-15 K=0..7,16..23; lanes 16-31 K=8..15,24..31
        const int ka = 2 * v + hf + 8 * hi + ((v >= 4) ? 8 : 0);
        a[e] = (__bf16)sA[lo * 33 + ka];
        // B 32x16 bf16 layout: lanes 0-15 K=0..7,16..23; lanes 16-31 K=8..15,24..31
        const int kb = kt + 2 * v + hf + 8 * hi + ((v >= 4) ? 8 : 0);
        b0[e] = (__bf16)B0[(size_t)kb * HC + n0 + lo];
        b1[e] = (__bf16)B1[(size_t)kb * HC + n0 + lo];
      }
    }
    c0 = __builtin_amdgcn_wmma_f32_16x16x32_bf16(false, a, false, b0, (short)0, c0, false, false);
    c1 = __builtin_amdgcn_wmma_f32_16x16x32_bf16(false, a, false, b1, (short)0, c1, false, false);
  }

  const float bb0 = bias0[n0 + lo];
  const float bb1 = bias1[n0 + lo];
#pragma unroll
  for (int r = 0; r < 8; ++r) {
    const int row = m0 + r + 8 * hi;               // D layout: M = r + 8*hi, N = lo
    C0[(size_t)row * HC + n0 + lo] = c0[r] + bb0;
    C1[(size_t)row * HC + n0 + lo] = c1[r] + bb1;
  }
}

// ---------------- per-edge attention logits + segment max ----------------
// grid-stride: one wave per edge per iteration; We (16x256) staged in LDS once per block.
__global__ void edge_logits(const int* __restrict__ srcA, const int* __restrict__ dstA,
                            const float* __restrict__ edge_attr, const float* __restrict__ loop_attr,
                            const float* __restrict__ xl, const float* __restrict__ xr,
                            const float* __restrict__ We, const float* __restrict__ att,
                            float* __restrict__ logits, float* __restrict__ mbuf) {
  __shared__ float sWe[ED_ * HC];
  __shared__ float sAtt[HC];
  const int tid = threadIdx.x;
  for (int i = tid; i < ED_ * HC; i += 256) sWe[i] = We[i];
  if (tid < HC) sAtt[tid] = att[tid];
  __syncthreads();

  const int wv = tid >> 5, lane = tid & 31;
  const int stride = EDGE_BLOCKS * 8;              // total waves

  for (int e = blockIdx.x * 8 + wv; e < ETOT; e += stride) {
    int s_n, d_n; const float* eaP;
    if (e < EE) { s_n = srcA[e]; d_n = dstA[e]; eaP = edge_attr + (size_t)e * ED_; }
    else        { s_n = d_n = e - EE;           eaP = loop_attr + (size_t)(e - EE) * ED_; }

    // prefetch next edge's gather rows (L2 -> near cache) while we compute
    const int e2 = e + stride;
    if (e2 < ETOT && lane < 8) {
      const int s2 = (e2 < EE) ? srcA[e2] : (e2 - EE);
      const int d2 = (e2 < EE) ? dstA[e2] : (e2 - EE);
      __builtin_prefetch(xl + (size_t)s2 * HC + lane * 32, 0, 0);
      __builtin_prefetch(xr + (size_t)d2 * HC + lane * 32, 0, 0);
    }

    float eav[ED_];
#pragma unroll
    for (int d = 0; d < ED_; ++d) eav[d] = eaP[d]; // broadcast loads

    const float* xls = xl + (size_t)s_n * HC;
    const float* xrd = xr + (size_t)d_n * HC;

#pragma unroll
    for (int h = 0; h < HH; ++h) {
      const int col = h * CC + lane;
      float ev = 0.f;
#pragma unroll
      for (int d = 0; d < ED_; ++d) ev = fmaf(eav[d], sWe[d * HC + col], ev);
      float z = xls[col] + xrd[col] + ev;
      z = (z > 0.f) ? z : NEG_SLOPE * z;           // leaky relu
      float t = z * sAtt[col];
#pragma unroll
      for (int off = 16; off > 0; off >>= 1) t += __shfl_xor(t, off, 32);
      if (lane == 0) {
        logits[(size_t)e * HH + h] = t;
        atomicMaxFloat(&mbuf[(size_t)d_n * HH + h], t);
      }
    }
  }
}

// ---------------- exp(logit - max) + segment sum ----------------
__global__ void edge_exp(const int* __restrict__ dstA, float* __restrict__ logits,
                         const float* __restrict__ mbuf, float* __restrict__ sbuf) {
  int idx = blockIdx.x * blockDim.x + threadIdx.x;
  if (idx >= ETOT * HH) return;
  const int e = idx >> 3, h = idx & 7;
  const int d_n = (e < EE) ? dstA[e] : (e - EE);
  const float p = __expf(logits[idx] - mbuf[(size_t)d_n * HH + h]);
  logits[idx] = p;                                 // reuse buffer for p
  atomicAdd(&sbuf[(size_t)d_n * HH + h], p);
}

// ---------------- scatter alpha * xl[src] into acc[dst] ----------------
__global__ void edge_scatter(const int* __restrict__ srcA, const int* __restrict__ dstA,
                             const float* __restrict__ xl, const float* __restrict__ p,
                             const float* __restrict__ sbuf, float* __restrict__ acc) {
  const int wv = threadIdx.x >> 5, lane = threadIdx.x & 31;
  const int e = blockIdx.x * 8 + wv;
  if (e >= ETOT) return;
  int s_n, d_n;
  if (e < EE) { s_n = srcA[e]; d_n = dstA[e]; } else { s_n = d_n = e - EE; }
  const float* xls = xl + (size_t)s_n * HC;
  float* accd = acc + (size_t)d_n * HC;
#pragma unroll
  for (int h = 0; h < HH; ++h) {
    const float alpha = p[(size_t)e * HH + h] / (sbuf[(size_t)d_n * HH + h] + 1e-16f);
    atomicAdd(&accd[h * CC + lane], alpha * xls[h * CC + lane]);
  }
}

// ---------------- head-mean + bias (+ optional ReLU+LayerNorm) ----------------
// one wave per node, lane = channel (C==32==wave32).
__global__ void node_final(const float* __restrict__ acc, const float* __restrict__ bias,
                           const float* __restrict__ g, const float* __restrict__ b,
                           float* __restrict__ out, int do_ln) {
  const int wv = threadIdx.x >> 5, lane = threadIdx.x & 31;
  const int n = blockIdx.x * 8 + wv;
  if (n >= NN) return;
  const float* a = acc + (size_t)n * HC;
  float v = 0.f;
#pragma unroll
  for (int h = 0; h < HH; ++h) v += a[h * CC + lane];
  v = v * (1.0f / HH) + bias[lane];
  if (do_ln) {
    v = fmaxf(v, 0.f);                             // relu
    float mu = v;
#pragma unroll
    for (int off = 16; off > 0; off >>= 1) mu += __shfl_xor(mu, off, 32);
    mu *= (1.0f / CC);
    float d = v - mu;
    float var = d * d;
#pragma unroll
    for (int off = 16; off > 0; off >>= 1) var += __shfl_xor(var, off, 32);
    var *= (1.0f / CC);
    v = d * rsqrtf(var + LN_EPS) * g[lane] + b[lane];
  }
  out[(size_t)n * CC + lane] = v;
}

// ---------------- launcher ----------------
extern "C" void kernel_launch(void* const* d_in, const int* in_sizes, int n_in,
                              void* d_out, int out_size, void* d_ws, size_t ws_size,
                              hipStream_t stream) {
  // setup_inputs() dict order
  const float* x     = (const float*)d_in[0];
  const int*   ei    = (const int*)  d_in[1];
  const float* ea    = (const float*)d_in[2];
  const float* Wl1   = (const float*)d_in[3];
  const float* bl1   = (const float*)d_in[4];
  const float* Wr1   = (const float*)d_in[5];
  const float* br1   = (const float*)d_in[6];
  const float* We1   = (const float*)d_in[7];
  const float* att1  = (const float*)d_in[8];
  const float* bias1 = (const float*)d_in[9];
  const float* Wl2   = (const float*)d_in[10];
  const float* bl2   = (const float*)d_in[11];
  const float* Wr2   = (const float*)d_in[12];
  const float* br2   = (const float*)d_in[13];
  const float* We2   = (const float*)d_in[14];
  const float* att2  = (const float*)d_in[15];
  const float* bias2 = (const float*)d_in[16];
  const float* ln_g  = (const float*)d_in[17];
  const float* ln_b  = (const float*)d_in[18];

  const int* srcA = ei;
  const int* dstA = ei + EE;

  // workspace carve-out
  char* w = (char*)d_ws;
  auto carve = [&](size_t bytes) { void* p = (void*)w; w += (bytes + 255) & ~(size_t)255; return p; };
  float* deg  = (float*)carve((size_t)NN * 4);
  float* lat  = (float*)carve((size_t)NN * ED_ * 4);       // self-loop attrs
  float* xl   = (float*)carve((size_t)NN * HC * 4);
  float* xr   = (float*)carve((size_t)NN * HC * 4);
  float* lg   = (float*)carve((size_t)ETOT * HH * 4);      // logits -> p
  float* mbuf = (float*)carve((size_t)NN * HH * 4);
  float* sbuf = (float*)carve((size_t)NN * HH * 4);
  float* acc  = (float*)carve((size_t)NN * HC * 4);
  float* h1   = (float*)carve((size_t)NN * CC * 4);
  float* out  = (float*)d_out;

  const dim3 blk(256);
  const int gEdgeWave = (ETOT + 7) / 8;                    // 1 wave / edge, 8 waves / block
  const int gNodeWave = (NN + 7) / 8;
  const dim3 gGemm(HC / 128, NN / 16);                     // (2, 3125)

  // --- self-loop attrs ---
  fillk<<<(NN + 255) / 256, blk, 0, stream>>>(deg, 0.f, NN);
  fillk<<<(NN * ED_ + 255) / 256, blk, 0, stream>>>(lat, 0.f, NN * ED_);
  edge_accum<<<(EE + 255) / 256, blk, 0, stream>>>(dstA, ea, deg, lat);
  loop_norm<<<(NN * ED_ + 255) / 256, blk, 0, stream>>>(lat, deg);

  // ================= layer 1 =================
  wmma_dual_gemm<<<gGemm, blk, 0, stream>>>(x, INF_, Wl1, bl1, xl, Wr1, br1, xr);
  fillk<<<(NN * HH + 255) / 256, blk, 0, stream>>>(mbuf, -3.0e38f, NN * HH);
  fillk<<<(NN * HH + 255) / 256, blk, 0, stream>>>(sbuf, 0.f, NN * HH);
  fillk<<<(NN * HC + 255) / 256, blk, 0, stream>>>(acc, 0.f, NN * HC);
  edge_logits<<<EDGE_BLOCKS, blk, 0, stream>>>(srcA, dstA, ea, lat, xl, xr, We1, att1, lg, mbuf);
  edge_exp<<<(ETOT * HH + 255) / 256, blk, 0, stream>>>(dstA, lg, mbuf, sbuf);
  edge_scatter<<<gEdgeWave, blk, 0, stream>>>(srcA, dstA, xl, lg, sbuf, acc);
  node_final<<<gNodeWave, blk, 0, stream>>>(acc, bias1, ln_g, ln_b, h1, 1);

  // ================= layer 2 =================
  wmma_dual_gemm<<<gGemm, blk, 0, stream>>>(h1, CC, Wl2, bl2, xl, Wr2, br2, xr);
  fillk<<<(NN * HH + 255) / 256, blk, 0, stream>>>(mbuf, -3.0e38f, NN * HH);
  fillk<<<(NN * HH + 255) / 256, blk, 0, stream>>>(sbuf, 0.f, NN * HH);
  fillk<<<(NN * HC + 255) / 256, blk, 0, stream>>>(acc, 0.f, NN * HC);
  edge_logits<<<EDGE_BLOCKS, blk, 0, stream>>>(srcA, dstA, ea, lat, xl, xr, We2, att2, lg, mbuf);
  edge_exp<<<(ETOT * HH + 255) / 256, blk, 0, stream>>>(dstA, lg, mbuf, sbuf);
  edge_scatter<<<gEdgeWave, blk, 0, stream>>>(srcA, dstA, xl, lg, sbuf, acc);
  node_final<<<gNodeWave, blk, 0, stream>>>(acc, bias2, ln_g, ln_b, out, 0);

  (void)in_sizes; (void)n_in; (void)out_size; (void)ws_size;
}
